// HypergraphConv_17111149707406
// MI455X (gfx1250) — matrix operand order repaired and checked by reading, not compiled, not run on hardware
//
#include <hip/hip_runtime.h>
#include <stdint.h>

// ---- problem constants (match reference) ----
#define NN 8192      // nodes
#define NE 4096      // hyperedges
#define CH 256       // channels (in == out)
#define EPS 1e-6f

typedef __attribute__((ext_vector_type(16))) __bf16        v16bf;
typedef __attribute__((ext_vector_type(8)))  float         v8f;
typedef __attribute__((ext_vector_type(8)))  unsigned int  v8u;

__device__ __forceinline__ unsigned short f2bf(float f) {
    union { float f; unsigned int u; } v; v.f = f;
    unsigned int r = v.u + 0x7FFFu + ((v.u >> 16) & 1u);   // RNE
    return (unsigned short)(r >> 16);
}
__device__ __forceinline__ float bf2f(unsigned short h) {
    union { unsigned int u; float f; } v; v.u = ((unsigned int)h) << 16;
    return v.f;
}

// LDS apertures are 4GB-aligned: low 32 bits of a generic pointer into LDS are
// the wave-relative LDS byte offset (ISA 10.2).
__device__ __forceinline__ unsigned lds_off(const void* p) {
    return (unsigned)(uintptr_t)p;
}

// CDNA5 async global->LDS copy (16B per lane), tracked by ASYNCcnt.
__device__ __forceinline__ void async_copy_b128(unsigned ldsDst, const void* gsrc) {
    asm volatile("global_load_async_to_lds_b128 %0, %1, off"
                 :: "v"(ldsDst), "v"(gsrc) : "memory");
}
__device__ __forceinline__ void wait_async0() {
    asm volatile("s_wait_asynccnt 0x0" ::: "memory");
}

// ---------------- pass 1: row sums of H (f32) + bf16 pack of H ----------------
__global__ __launch_bounds__(256)
void rowsum_pack_kernel(const float* __restrict__ H,
                        unsigned short* __restrict__ Hbf,
                        float* __restrict__ d_v) {
    __shared__ float red[256];
    const int r = blockIdx.x;
    const int tid = threadIdx.x;
    const float* row = H + (size_t)r * NE;
    unsigned short* rowb = Hbf + (size_t)r * NE;
    float s = 0.f;
#pragma unroll
    for (int j = 0; j < NE / 256; ++j) {
        int c = tid + j * 256;
        float v = row[c];
        rowb[c] = f2bf(v);
        s += v;
    }
    red[tid] = s; __syncthreads();
    for (int off = 128; off > 0; off >>= 1) {
        if (tid < off) red[tid] += red[tid + off];
        __syncthreads();
    }
    if (tid == 0) d_v[r] = red[0];
}

// ---------------- pass 2: column sums (deterministic two-stage) ----------------
__global__ __launch_bounds__(256)
void colsum_partial_kernel(const unsigned short* __restrict__ Hbf,
                           float* __restrict__ partial) {
    const int c = blockIdx.x * 256 + threadIdx.x;  // 0..NE-1
    const int chunk = blockIdx.y;                  // 0..7, 1024 rows each
    float s = 0.f;
    const int r0 = chunk * (NN / 8);
    for (int r = r0; r < r0 + NN / 8; ++r)
        s += bf2f(Hbf[(size_t)r * NE + c]);
    partial[(size_t)chunk * NE + c] = s;
}

// ---------------- scales: dv^{-1/2}, de^{-1} ----------------
__global__ __launch_bounds__(256)
void scales_kernel(const float* __restrict__ d_v,
                   const float* __restrict__ partial,
                   float* __restrict__ dv_is,
                   float* __restrict__ de_inv) {
    const int i = blockIdx.x * 256 + threadIdx.x;  // up to NN
    if (i < NN) dv_is[i] = rsqrtf(d_v[i] + EPS);
    if (i < NE) {
        float s = 0.f;
#pragma unroll
        for (int j = 0; j < 8; ++j) s += partial[(size_t)j * NE + i];
        de_inv[i] = 1.f / (s + EPS);
    }
}

// ---------------- xs = dv^{-1/2} * x, packed bf16 ----------------
__global__ __launch_bounds__(256)
void xs_kernel(const float* __restrict__ x, const float* __restrict__ dv_is,
               unsigned short* __restrict__ xs) {
    const int i = blockIdx.x * 256 + threadIdx.x;  // over NN*CH
    const int r = i >> 8;                          // CH == 256
    xs[i] = f2bf(x[i] * dv_is[r]);
}

// ---------------- Wt[c][o] = W[o][c], packed bf16 ----------------
__global__ __launch_bounds__(256)
void wt_kernel(const float* __restrict__ W, unsigned short* __restrict__ Wt) {
    const int i = blockIdx.x * 256 + threadIdx.x;  // over CH*CH
    const int o = i >> 8, c = i & 255;
    Wt[(size_t)c * CH + o] = f2bf(W[i]);
}

// ---------------- LDS-tiled bf16 transpose: Hbf[N][E] -> HbfT[E][N] ----------------
__global__ __launch_bounds__(256)
void transpose_bf_kernel(const unsigned short* __restrict__ src,
                         unsigned short* __restrict__ dst) {
    __shared__ unsigned short tile[32][33];
    const int tx = threadIdx.x & 31, ty0 = threadIdx.x >> 5;
    const int r0 = blockIdx.y * 32;  // N dim
    const int c0 = blockIdx.x * 32;  // E dim
#pragma unroll
    for (int j = 0; j < 4; ++j) {
        int ty = ty0 + 8 * j;
        tile[ty][tx] = src[(size_t)(r0 + ty) * NE + c0 + tx];
    }
    __syncthreads();
#pragma unroll
    for (int j = 0; j < 4; ++j) {
        int ty = ty0 + 8 * j;
        dst[(size_t)(c0 + ty) * NN + r0 + tx] = tile[tx][ty];
    }
}

// ---------------- generic bf16 WMMA GEMM, async-pipelined, fused epilogues ----------------
// C[M,Nc] = A[M,K] * B[K,Nc], f32 accumulate.
// MODE 0: out_bf[r][c] = bf16( acc * scaleRow[r] )
// MODE 2: out_f [r][c] = acc + bias[c]
#define BM 64
#define BN 64
#define BK 32
#define LDK 34   // padded LDS row stride (elems) -> 17 dwords, coprime with 64 banks

template <int MODE>
__global__ __launch_bounds__(128)
void wmma_gemm_kernel(const unsigned short* __restrict__ A,
                      const unsigned short* __restrict__ B,
                      int Nc, int K,
                      const float* __restrict__ scaleRow,
                      const float* __restrict__ bias,
                      unsigned short* __restrict__ outBf,
                      float* __restrict__ outF) {
    __shared__ unsigned short sA[2][BM][LDK];   // [buf][m][k]
    __shared__ unsigned short sB[2][BN][LDK];   // [buf][n][k] (transposed)

    const int tid  = threadIdx.x;
    const int lane = tid & 31;               // wave32
    const int wave = tid >> 5;               // 4 waves
    const int wm   = (wave >> 1) * 32;
    const int wn   = (wave & 1) * 32;
    const unsigned rowBlk = blockIdx.x * BM;
    const unsigned colBlk = blockIdx.y * BN;

    v8f acc[2][2] = {};

    const int lrow   = lane & 15;
    const int khalf  = (lane >> 4) << 3;     // A: 0 or 8
    const int kbhalf = (lane >> 4) << 4;     // B: 0 or 16

    // static per-thread tile-staging coordinates
    const int rA = tid >> 2;             // 0..31 (A rows, two halves of 32)
    const int cA = (tid & 3) * 8;        // 0,8,16,24
    const int kB = tid >> 3;             // 0..15 (B rows, two halves of 16)
    const int nB = (tid & 7) * 8;        // 0..56

    uint4 breg[2];

    // async copy of the 64x32 A tile (16B per lane per half) into LDS buffer `buf`
    auto issueA = [&](unsigned kt, int buf) {
#pragma unroll
        for (int half = 0; half < 2; ++half) {
            const int r = rA + half * 32;
            const unsigned short* g = A + (unsigned)(rowBlk + r) * (unsigned)K + kt + cA;
            async_copy_b128(lds_off(&sA[buf][r][cA]), g);
        }
    };
    auto loadB = [&](unsigned kt) {
#pragma unroll
        for (int half = 0; half < 2; ++half)
            breg[half] = *(const uint4*)(B + (unsigned)(kt + kB + half * 16) * (unsigned)Nc + colBlk + nB);
    };
    auto storeB = [&](int buf) {
#pragma unroll
        for (int half = 0; half < 2; ++half) {
            const int k = kB + half * 16;
            const uint4 d = breg[half];
            sB[buf][nB + 0][k] = (unsigned short)(d.x);
            sB[buf][nB + 1][k] = (unsigned short)(d.x >> 16);
            sB[buf][nB + 2][k] = (unsigned short)(d.y);
            sB[buf][nB + 3][k] = (unsigned short)(d.y >> 16);
            sB[buf][nB + 4][k] = (unsigned short)(d.z);
            sB[buf][nB + 5][k] = (unsigned short)(d.z >> 16);
            sB[buf][nB + 6][k] = (unsigned short)(d.w);
            sB[buf][nB + 7][k] = (unsigned short)(d.w >> 16);
        }
    };

    // preamble: fill buffer 0
    issueA(0, 0);
    loadB(0);
    storeB(0);
    wait_async0();
    __syncthreads();

    const int nIter = K / BK;
    for (int it = 0; it < nIter; ++it) {
        const int p = it & 1;
        const bool more = (it + 1) < nIter;
        if (more) {                       // prefetch next tile while computing
            issueA((unsigned)(it + 1) * BK, p ^ 1);
            loadB((unsigned)(it + 1) * BK);
        }

        // fragments per ISA 7.12.2 bf16 layouts (packed K-pairs -> one b32 read each)
        v8u ua0, ua1, ub0, ub1;
#pragma unroll
        for (int v = 0; v < 8; ++v) {
            const int kA = (v < 4) ? (khalf + 2 * v) : (16 + khalf + 2 * (v - 4));
            ua0[v] = *(const unsigned int*)&sA[p][wm + lrow][kA];
            ua1[v] = *(const unsigned int*)&sA[p][wm + 16 + lrow][kA];
            const int kBf = (v < 4) ? (khalf + 2 * v) : (16 + khalf + 2 * (v - 4));
            ub0[v] = *(const unsigned int*)&sB[p][wn + lrow][kBf];
            ub1[v] = *(const unsigned int*)&sB[p][wn + 16 + lrow][kBf];
        }
        const v16bf a0 = __builtin_bit_cast(v16bf, ua0);
        const v16bf a1 = __builtin_bit_cast(v16bf, ua1);
        const v16bf b0 = __builtin_bit_cast(v16bf, ub0);
        const v16bf b1 = __builtin_bit_cast(v16bf, ub1);

        acc[0][0] = __builtin_amdgcn_wmma_f32_16x16x32_bf16(false, a0, false, b0, (short)0, acc[0][0], false, false);
        acc[0][1] = __builtin_amdgcn_wmma_f32_16x16x32_bf16(false, a0, false, b1, (short)0, acc[0][1], false, false);
        acc[1][0] = __builtin_amdgcn_wmma_f32_16x16x32_bf16(false, a1, false, b0, (short)0, acc[1][0], false, false);
        acc[1][1] = __builtin_amdgcn_wmma_f32_16x16x32_bf16(false, a1, false, b1, (short)0, acc[1][1], false, false);

        if (more) storeB(p ^ 1);          // scatter-transpose B into the next buffer
        wait_async0();                    // A-tile async copy for next buffer done
        __syncthreads();                  // all waves' B stores visible
    }

    // Epilogue: C/D layout — VGPR r: lanes 0-15 -> (M=r, N=lane), 16-31 -> (M=8+r, N=lane-16)
    const int colLane = lane & 15;
    const int rowHalf = (lane >> 4) * 8;
#pragma unroll
    for (int i = 0; i < 2; ++i)
#pragma unroll
        for (int j = 0; j < 2; ++j)
#pragma unroll
            for (int r = 0; r < 8; ++r) {
                const unsigned gr = rowBlk + wm + i * 16 + rowHalf + r;
                const unsigned gc = colBlk + wn + j * 16 + colLane;
                const float v = acc[i][j][r];
                if (MODE == 2)
                    outF[gr * (unsigned)Nc + gc] = v + bias[gc];
                else
                    outBf[gr * (unsigned)Nc + gc] = f2bf(v * scaleRow[gr]);
            }
}

// ---------------- host-side orchestration ----------------
extern "C" void kernel_launch(void* const* d_in, const int* in_sizes, int n_in,
                              void* d_out, int out_size, void* d_ws, size_t ws_size,
                              hipStream_t stream) {
    const float* x = (const float*)d_in[0];   // [NN, CH]
    const float* H = (const float*)d_in[1];   // [NN, NE]
    const float* W = (const float*)d_in[2];   // [CH, CH]
    const float* b = (const float*)d_in[3];   // [CH]
    float* out = (float*)d_out;               // [NN, CH] f32

    char* ws = (char*)d_ws;
    auto carve = [&](size_t bytes) {
        char* p = ws;
        ws += (bytes + 255) & ~size_t(255);
        return p;
    };
    float* d_v      = (float*)carve((size_t)NN * 4);
    float* partial  = (float*)carve((size_t)8 * NE * 4);
    float* dv_is    = (float*)carve((size_t)NN * 4);
    float* de_inv   = (float*)carve((size_t)NE * 4);
    unsigned short* Hbf  = (unsigned short*)carve((size_t)NN * NE * 2);
    unsigned short* HbfT = (unsigned short*)carve((size_t)NN * NE * 2);
    unsigned short* xs   = (unsigned short*)carve((size_t)NN * CH * 2);
    unsigned short* tbf  = (unsigned short*)carve((size_t)NE * CH * 2);
    unsigned short* ybf  = (unsigned short*)carve((size_t)NN * CH * 2);
    unsigned short* Wt   = (unsigned short*)carve((size_t)CH * CH * 2);

    rowsum_pack_kernel<<<NN, 256, 0, stream>>>(H, Hbf, d_v);
    {
        dim3 g(NE / 256, 8);
        colsum_partial_kernel<<<g, 256, 0, stream>>>(Hbf, partial);
    }
    scales_kernel<<<NN / 256, 256, 0, stream>>>(d_v, partial, dv_is, de_inv);
    xs_kernel<<<(NN * CH) / 256, 256, 0, stream>>>(x, dv_is, xs);
    wt_kernel<<<(CH * CH) / 256, 256, 0, stream>>>(W, Wt);
    {
        dim3 g(NE / 32, NN / 32);
        transpose_bf_kernel<<<g, 256, 0, stream>>>(Hbf, HbfT);
    }
    // GEMM1: t = de_inv ⊙ (H^T @ xs)   A=HbfT [NE x NN], B=xs [NN x CH]
    {
        dim3 g(NE / BM, CH / BN);
        wmma_gemm_kernel<0><<<g, 128, 0, stream>>>(HbfT, xs, CH, NN, de_inv, nullptr, tbf, nullptr);
    }
    // GEMM2: y = dv_is ⊙ (H @ t)       A=Hbf [NN x NE], B=tbf [NE x CH]
    {
        dim3 g(NN / BM, CH / BN);
        wmma_gemm_kernel<0><<<g, 128, 0, stream>>>(Hbf, tbf, CH, NE, dv_is, nullptr, ybf, nullptr);
    }
    // GEMM3: out = y @ W^T + b         A=ybf [NN x CH], B=Wt [CH x CH]
    {
        dim3 g(NN / BM, CH / BN);
        wmma_gemm_kernel<2><<<g, 128, 0, stream>>>(ybf, Wt, CH, CH, nullptr, b, nullptr, out);
    }
}
